// CausalAttention_3848290697375
// MI455X (gfx1250) — matrix operand order repaired and checked by reading, not compiled, not run on hardware
//
#include <hip/hip_runtime.h>
#include <math.h>

// ---------------------------------------------------------------------------
// CausalAttention for MI455X (gfx1250, wave32, WMMA).
//
// Roofline: ~0.24 TFLOP, ~220MB unique traffic; fits in 192MB L2, HBM 23.3TB/s
// -> matrix-pipe-bound. All four GEMMs (xW x3, QK^T, PV) go through
// v_wmma_f32_16x16x32_bf16 (the high-throughput bf16 K=32 shape; f32 WMMA
// only exists at K=4).
//
// Pipeline (all via d_ws, bf16 intermediates):
//   kernel 0a: x  -> xbf            (one-time convert, hoists cvt off hot loop)
//   kernel 0b: Wq/Wk/Wv -> Wt bf16, TRANSPOSED (B-frag loads become contiguous)
//   kernel 1 : Q,K (row-major) and V TRANSPOSED = xbf @ W
//   kernel 2 : flash attention; 8-wave block = one 16-row q-tile; D split
//              128-wide per wave; Q frags held in registers across the key
//              loop; S reduced via LDS; P restaged via padded per-wave LDS.
// ---------------------------------------------------------------------------

#define D_DIM 1024
#define BATCH 8
#define SEQ   2048
#define NTOK  (BATCH * SEQ)         // 16384 rows
#define NELEM ((size_t)NTOK * D_DIM) // 16,777,216

typedef __bf16 bf16_t;
typedef __attribute__((ext_vector_type(8)))  __bf16 v8bf;
typedef __attribute__((ext_vector_type(16))) __bf16 v16bf;
typedef __attribute__((ext_vector_type(8)))  float  v8f;

__device__ inline bf16_t f2bf(float f) {
  // round-to-nearest-even f32 -> bf16 (bit-exact, no cvt-selection risk)
  unsigned u = __builtin_bit_cast(unsigned, f);
  unsigned r = (u + 0x7FFFu + ((u >> 16) & 1u)) >> 16;
  return __builtin_bit_cast(bf16_t, (unsigned short)r);
}

__device__ inline v16bf cat8(v8bf a, v8bf b) {
  return __builtin_shufflevector(a, b, 0,1,2,3,4,5,6,7,8,9,10,11,12,13,14,15);
}

// ---------------------------------------------------------------------------
// Kernel 0a: x (f32) -> xbf (bf16), one-time, bandwidth-trivial (96MB total)
// ---------------------------------------------------------------------------
__global__ __launch_bounds__(256)
void cvt_x_kernel(const float* __restrict__ x, bf16_t* __restrict__ xb) {
  const size_t i = ((size_t)blockIdx.x * 256 + threadIdx.x) * 4;
  float4 v = *(const float4*)(x + i);
  xb[i + 0] = f2bf(v.x);
  xb[i + 1] = f2bf(v.y);
  xb[i + 2] = f2bf(v.z);
  xb[i + 3] = f2bf(v.w);
}

// ---------------------------------------------------------------------------
// Kernel 0b: W (d_in,d_out f32) -> Wt (3,d_out,d_in bf16), transposed so the
// projection B-fragment is a contiguous 32B run per lane.
// ---------------------------------------------------------------------------
__global__ __launch_bounds__(256)
void cvt_w_kernel(const float* __restrict__ Wq, const float* __restrict__ Wk,
                  const float* __restrict__ Wv, bf16_t* __restrict__ wt) {
  const int i   = blockIdx.x * 256 + threadIdx.x;   // 0 .. 3*D*D-1
  const int mat = i / (D_DIM * D_DIM);
  const int rem = i % (D_DIM * D_DIM);
  const int k   = rem / D_DIM;
  const int n   = rem % D_DIM;
  const float* W = (mat == 0) ? Wq : (mat == 1) ? Wk : Wv;
  wt[(size_t)mat * D_DIM * D_DIM + (size_t)n * D_DIM + k] = f2bf(W[(size_t)k * D_DIM + n]);
}

// ---------------------------------------------------------------------------
// Kernel 1: projections. One wave = one 16x16 tile of Q, K, or V.
// Q,K stored [B*T, D]; V stored TRANSPOSED [B, D, T] for the attention
// B-fragment. grid = (B*T/16, 192/8), block = 256.
// ---------------------------------------------------------------------------
__global__ __launch_bounds__(256)
void qkv_proj_kernel(const bf16_t* __restrict__ xb,
                     const bf16_t* __restrict__ wt,
                     bf16_t* __restrict__ q_out,
                     bf16_t* __restrict__ k_out,
                     bf16_t* __restrict__ vt_out) {
  const int lane = threadIdx.x & 31;
  const int wave = threadIdx.x >> 5;
  const int hi   = lane >> 4;
  const int ln   = lane & 15;

  const int mtile = blockIdx.x;
  const int ntile = blockIdx.y * 8 + wave;     // 0..191 over {Q,K,V} x 64
  const int mat   = ntile >> 6;
  const int ncol  = (ntile & 63) << 4;

  const bf16_t* xrow = xb + (size_t)(mtile * 16 + ln) * D_DIM;
  const bf16_t* wrow = wt + ((size_t)mat * D_DIM + ncol + ln) * D_DIM;

  v8f acc = {};
  for (int k0 = 0; k0 < D_DIM; k0 += 32) {
    // A frag: lane = row; elem e -> k = k0 + e + hi*16 (contiguous)
    v16bf a = cat8(*(const v8bf*)(xrow + k0 + (hi ? 16 : 0)),
                   *(const v8bf*)(xrow + k0 + (hi ? 16 : 0) + 8));
    // B frag: lane = col n; elem e -> k = k0 + e + hi*16 (contiguous in Wt)
    v16bf b = cat8(*(const v8bf*)(wrow + k0 + (hi ? 16 : 0)),
                   *(const v8bf*)(wrow + k0 + (hi ? 16 : 0) + 8));
    acc = __builtin_amdgcn_wmma_f32_16x16x32_bf16(false, a, false, b,
                                                  (short)0, acc, false, false);
  }
  // C layout: VGPR r -> row r + hi*8, lane ln -> col ncol+ln
  if (mat < 2) {
    bf16_t* out = (mat == 0) ? q_out : k_out;
#pragma unroll
    for (int r = 0; r < 8; ++r) {
      const int row = mtile * 16 + r + hi * 8;
      out[(size_t)row * D_DIM + ncol + ln] = f2bf(acc[r]);
    }
  } else {
    // V transposed: vt[b][d][t]
#pragma unroll
    for (int r = 0; r < 8; ++r) {
      const int g = mtile * 16 + r + hi * 8;   // global token
      const int bb = g >> 11, t = g & (SEQ - 1);
      vt_out[((size_t)bb * D_DIM + ncol + ln) * SEQ + t] = f2bf(acc[r]);
    }
  }
}

// ---------------------------------------------------------------------------
// Kernel 2: flash attention. block = 8 waves = one (batch, 16-row q-tile).
// Each wave owns a 128-wide D slice for both the S K-split and the O columns;
// its 4 Q A-fragments live in registers for the whole key loop.
// grid = B * T/16 = 1024, block = 256.
// ---------------------------------------------------------------------------
#define PSTRIDE 40   // padded row stride for P staging (bank-conflict-free)

__global__ __launch_bounds__(256)
void flash_attn_kernel(const bf16_t* __restrict__ Q,
                       const bf16_t* __restrict__ K,
                       const bf16_t* __restrict__ Vt,
                       float* __restrict__ out) {
  __shared__ float  sred[8][512];           // per-wave partial S (16x32)  16KB
  __shared__ bf16_t plds[8][16 * PSTRIDE];  // per-wave P staging          10KB

  const int lane = threadIdx.x & 31;
  const int wave = threadIdx.x >> 5;
  const int hi   = lane >> 4;
  const int ln   = lane & 15;

  const int b  = blockIdx.x >> 7;
  const int q0 = (blockIdx.x & 127) << 4;

  const bf16_t* Qb  = Q  + (size_t)b * SEQ * D_DIM;
  const bf16_t* Kb  = K  + (size_t)b * SEQ * D_DIM;
  const bf16_t* Vtb = Vt + (size_t)b * D_DIM * SEQ;

  const int   colbase = wave * 128;
  const float scale   = 0.03125f;  // 1/sqrt(1024)

  // Preload this wave's 4 Q A-fragments (16 rows x 128 d-slice) into VGPRs
  const bf16_t* qrow = Qb + (size_t)(q0 + ln) * D_DIM + colbase;
  v16bf qa[4];
#pragma unroll
  for (int dd = 0; dd < 4; ++dd) {
    qa[dd] = cat8(*(const v8bf*)(qrow + dd * 32 + (hi ? 16 : 0)),
                  *(const v8bf*)(qrow + dd * 32 + (hi ? 16 : 0) + 8));
  }

  v8f o[8];
#pragma unroll
  for (int t = 0; t < 8; ++t) o[t] = (v8f){};
  float mrow[8], lrow[8];
#pragma unroll
  for (int r = 0; r < 8; ++r) { mrow[r] = -__builtin_inff(); lrow[r] = 0.f; }

  for (int k0 = 0; k0 <= q0 + 15; k0 += 32) {   // 32-key causal chunks
    // --- partial S = Q·K^T over this wave's 128-wide d-slice (8 WMMAs) ---
    v8f s0 = {}, s1 = {};
#pragma unroll
    for (int dd = 0; dd < 4; ++dd) {
      const int d0 = colbase + dd * 32;
      // K^T as B frag: lane = key, elems = contiguous d run of 16
      const bf16_t* kr0 = Kb + (size_t)(k0 + ln)      * D_DIM + d0 + (hi ? 16 : 0);
      const bf16_t* kr1 = Kb + (size_t)(k0 + 16 + ln) * D_DIM + d0 + (hi ? 16 : 0);
      v16bf b0 = cat8(*(const v8bf*)kr0, *(const v8bf*)(kr0 + 8));
      v16bf b1 = cat8(*(const v8bf*)kr1, *(const v8bf*)(kr1 + 8));
      s0 = __builtin_amdgcn_wmma_f32_16x16x32_bf16(false, qa[dd], false, b0, (short)0, s0, false, false);
      s1 = __builtin_amdgcn_wmma_f32_16x16x32_bf16(false, qa[dd], false, b1, (short)0, s1, false, false);
    }

    // --- reduce the 8 K-split partials through LDS ---
#pragma unroll
    for (int e = 0; e < 8; ++e) {
      sred[wave][e * 32 + lane]       = s0[e];
      sred[wave][(e + 8) * 32 + lane] = s1[e];
    }
    __syncthreads();
    float S0[8], S1[8];
#pragma unroll
    for (int e = 0; e < 8; ++e) { S0[e] = 0.f; S1[e] = 0.f; }
#pragma unroll
    for (int w = 0; w < 8; ++w) {
#pragma unroll
      for (int e = 0; e < 8; ++e) {
        S0[e] += sred[w][e * 32 + lane];
        S1[e] += sred[w][(e + 8) * 32 + lane];
      }
    }
    __syncthreads();

    // --- causal mask + online softmax (identical across waves) ---
    const int col0 = k0 + ln, col1 = col0 + 16;
    float c[8];
#pragma unroll
    for (int r = 0; r < 8; ++r) {
      const int row = q0 + r + hi * 8;
      float l0 = (col0 <= row) ? S0[r] * scale : -__builtin_inff();
      float l1 = (col1 <= row) ? S1[r] * scale : -__builtin_inff();
      float e = fmaxf(l0, l1);
      e = fmaxf(e, __shfl_xor(e, 1));
      e = fmaxf(e, __shfl_xor(e, 2));
      e = fmaxf(e, __shfl_xor(e, 4));
      e = fmaxf(e, __shfl_xor(e, 8));   // stays within 16-lane half (wave32)
      const float mnew = fmaxf(mrow[r], e);
      const float p0 = __expf(l0 - mnew);
      const float p1 = __expf(l1 - mnew);
      float rs = p0 + p1;
      rs += __shfl_xor(rs, 1);
      rs += __shfl_xor(rs, 2);
      rs += __shfl_xor(rs, 4);
      rs += __shfl_xor(rs, 8);
      const float cr = __expf(mrow[r] - mnew);
      lrow[r] = lrow[r] * cr + rs;
      mrow[r] = mnew;
      c[r] = cr;
      // stash P (16x32 bf16, padded rows) into this wave's LDS region
      plds[wave][(r + hi * 8) * PSTRIDE + ln]      = f2bf(p0);
      plds[wave][(r + hi * 8) * PSTRIDE + ln + 16] = f2bf(p1);
    }
    // rescale O accumulators
#pragma unroll
    for (int t = 0; t < 8; ++t)
#pragma unroll
      for (int r = 0; r < 8; ++r) o[t][r] *= c[r];

    // P as A frag (same-wave LDS round trip; DS ops in-order within wave)
    v16bf pa;
#pragma unroll
    for (int e = 0; e < 8; ++e) {
      pa[e]     = plds[wave][ln * PSTRIDE + e + (hi ? 16 : 0)];
      pa[e + 8] = plds[wave][ln * PSTRIDE + 8 + e + (hi ? 16 : 0)];
    }

    // --- O += P · V over this wave's 128-col slice (8 WMMAs) ---
    // Vt[b][col][key]: B-frag elem e -> key = k0 + e + hi*16, contiguous run
#pragma unroll
    for (int t = 0; t < 8; ++t) {
      const bf16_t* vr = Vtb + (size_t)(colbase + t * 16 + ln) * SEQ + k0 + (hi ? 16 : 0);
      v16bf vb = cat8(*(const v8bf*)vr, *(const v8bf*)(vr + 8));
      o[t] = __builtin_amdgcn_wmma_f32_16x16x32_bf16(false, pa, false, vb,
                                                     (short)0, o[t], false, false);
    }
  }

  // --- normalize and store f32 output ---
#pragma unroll
  for (int r = 0; r < 8; ++r) {
    const float inv = 1.0f / lrow[r];
    const int row = q0 + r + hi * 8;
#pragma unroll
    for (int t = 0; t < 8; ++t) {
      out[((size_t)b * SEQ + row) * D_DIM + colbase + t * 16 + ln] = o[t][r] * inv;
    }
  }
}

// ---------------------------------------------------------------------------
extern "C" void kernel_launch(void* const* d_in, const int* in_sizes, int n_in,
                              void* d_out, int out_size, void* d_ws, size_t ws_size,
                              hipStream_t stream) {
  const float* x  = (const float*)d_in[0];
  const float* Wq = (const float*)d_in[1];
  const float* Wk = (const float*)d_in[2];
  const float* Wv = (const float*)d_in[3];
  float* out = (float*)d_out;

  // Workspace layout (bf16): Q | K | Vt | xbf | Wt   = 4*32MB + 6MB = 140MB
  bf16_t* qbf = (bf16_t*)d_ws;
  bf16_t* kbf = qbf + NELEM;
  bf16_t* vtb = kbf + NELEM;
  bf16_t* xbf = vtb + NELEM;
  bf16_t* wtb = xbf + NELEM;

  cvt_x_kernel<<<dim3(NELEM / 1024), 256, 0, stream>>>(x, xbf);
  cvt_w_kernel<<<dim3(3 * D_DIM * D_DIM / 256), 256, 0, stream>>>(Wq, Wk, Wv, wtb);

  dim3 grid1(NTOK / 16, (3 * D_DIM / 16) / 8);   // (1024, 24)
  qkv_proj_kernel<<<grid1, 256, 0, stream>>>(xbf, wtb, qbf, kbf, vtb);

  dim3 grid2(BATCH * (SEQ / 16));                // 1024
  flash_attn_kernel<<<grid2, 256, 0, stream>>>(qbf, kbf, vtb, out);
}
